// MultiHeadAttentionWithKVCache_33595234189986
// MI455X (gfx1250) — matrix operand order repaired and checked by reading, not compiled, not run on hardware
//
#include <hip/hip_runtime.h>
#include <hip/hip_bf16.h>

// ---------------------------------------------------------------------------
// Multi-head attention, bf16 WMMA pipeline for gfx1250 (wave32, 16x16x32 bf16)
//   1) fp32 -> bf16 conversion of x and the four weight matrices
//   2) Q/K/V projections: C[M,N] = A[M,K] * W[N,K]^T
//      128x128 block tile, 8 waves, double-buffered ASYNC global->LDS staging
//   3) flash attention per (b, h, 64-row q tile), online softmax, WMMA P*V
//   4) output projection (bf16 in, f32 out -> d_out)
// ---------------------------------------------------------------------------

typedef __bf16 bf16_t;
typedef __attribute__((ext_vector_type(16))) __bf16 v16bf;
typedef __attribute__((ext_vector_type(8)))  float  v8f;

__device__ __forceinline__ bf16_t f2bf(float f) {
  union { float f; unsigned u; } in; in.f = f;
  unsigned u = in.u;
  u += 0x7FFFu + ((u >> 16) & 1u);            // round-to-nearest-even
  union { unsigned short s; bf16_t b; } out;
  out.s = (unsigned short)(u >> 16);
  return out.b;
}

// CDNA5 async copy: global -> LDS, 16 bytes per lane, tracked by ASYNCcnt.
// GVS mode: mem_addr = SADDR(64b) + VADDR(32b byte offset). VDST holds the
// 32-bit LDS byte address (generic LDS pointer truncated to its low 32 bits).
__device__ __forceinline__ void async_load_b128(unsigned lds_addr, unsigned goff,
                                                unsigned long long base) {
  asm volatile("global_load_async_to_lds_b128 %0, %1, %2"
               :: "v"(lds_addr), "v"(goff), "s"(base)
               : "memory");
}

// A-matrix fragment (16x32 bf16, MxK): lane = (grp<<4)|m ; per lane two
// contiguous 8-half chunks: k = kbase+grp*8 .. +7  and  k = kbase+16+grp*8 .. +7
__device__ __forceinline__ v16bf load_a_frag(const bf16_t* row, int grp, int kbase) {
  union { v16bf v; bf16_t h[16]; } u;
  __builtin_memcpy(&u.h[0], row + kbase + grp * 8, 16);
  __builtin_memcpy(&u.h[8], row + kbase + 16 + grp * 8, 16);
  return u.v;
}

// B-matrix fragment (32x16 bf16, KxN): lane holds column n; two contiguous
// 8-half chunks: k = kbase+grp*8 .. +7  and  k = kbase+16+grp*8 .. +7
__device__ __forceinline__ v16bf load_b_frag(const bf16_t* row, int grp, int kbase) {
  union { v16bf v; bf16_t h[16]; } u;
  __builtin_memcpy(&u.h[0], row + kbase + grp * 8, 16);
  __builtin_memcpy(&u.h[8], row + kbase + 16 + grp * 8, 16);
  return u.v;
}

// ---------------------------------------------------------------------------
// fp32 -> bf16 conversion, 4 elements / thread
// ---------------------------------------------------------------------------
__global__ __launch_bounds__(256) void cvt_f32_bf16(const float* __restrict__ s,
                                                    bf16_t* __restrict__ d,
                                                    long n4) {
  long i = (long)blockIdx.x * blockDim.x + threadIdx.x;
  if (i >= n4) return;
  float4 f = *(const float4*)(s + i * 4);
  bf16_t o[4] = { f2bf(f.x), f2bf(f.y), f2bf(f.z), f2bf(f.w) };
  *(uint2*)(d + i * 4) = *(const uint2*)o;
}

// ---------------------------------------------------------------------------
// Tiled GEMM: C[M,N] = A[M,K] * B[N,K]^T, bf16 inputs, f32 accumulate.
// 256 threads = 8 waves; block tile 128x128, BK=64; wave tile 32x64
// (2 m-subtiles x 4 n-subtiles = 16 WMMAs per BK step per wave).
// Double-buffered LDS fed by global_load_async_to_lds_b128.
// ---------------------------------------------------------------------------
#define BM 128
#define BN 128
#define BK 64
#define LDT (BK + 8)   // halves; 144 B row stride (16B aligned)

template <int OUT_BF16>
__global__ __launch_bounds__(256) void gemm_abt(const bf16_t* __restrict__ A,
                                                const bf16_t* __restrict__ B,
                                                float* __restrict__ Cf,
                                                bf16_t* __restrict__ Cb,
                                                int M, int N, int K) {
  __shared__ bf16_t sA[2][BM * LDT];
  __shared__ bf16_t sB[2][BN * LDT];
  const int tid  = threadIdx.x;
  const int lane = tid & 31;
  const int wv   = tid >> 5;
  const int grp  = lane >> 4;
  const int ln   = lane & 15;
  const int rb   = wv >> 1;      // row band: 32 rows each
  const int cb   = wv & 1;       // col band: 64 cols each
  const int bm   = blockIdx.y;
  const int bn   = blockIdx.x;
  const unsigned long long baseA = (unsigned long long)A;
  const unsigned long long baseB = (unsigned long long)B;

  v8f acc[2][4] = {};

  const int nk = K / BK;
  // prologue: stage slab 0 into buffer 0 (1024 16B chunks per matrix, 4/thread)
#pragma unroll
  for (int j = 0; j < 4; ++j) {
    int chunk = j * 256 + tid;
    int row   = chunk >> 3;
    int c8    = (chunk & 7) << 3;
    unsigned la = (unsigned)(size_t)&sA[0][row * LDT + c8];
    unsigned lb = (unsigned)(size_t)&sB[0][row * LDT + c8];
    unsigned ga = (unsigned)(((size_t)(bm * BM + row) * K + c8) * 2);
    unsigned gb = (unsigned)(((size_t)(bn * BN + row) * K + c8) * 2);
    async_load_b128(la, ga, baseA);
    async_load_b128(lb, gb, baseB);
  }

  for (int s = 0; s < nk; ++s) {
    const int cur = s & 1;
    if (s + 1 < nk) {
      const int k0n = (s + 1) * BK;
#pragma unroll
      for (int j = 0; j < 4; ++j) {
        int chunk = j * 256 + tid;
        int row   = chunk >> 3;
        int c8    = (chunk & 7) << 3;
        unsigned la = (unsigned)(size_t)&sA[cur ^ 1][row * LDT + c8];
        unsigned lb = (unsigned)(size_t)&sB[cur ^ 1][row * LDT + c8];
        unsigned ga = (unsigned)(((size_t)(bm * BM + row) * K + k0n + c8) * 2);
        unsigned gb = (unsigned)(((size_t)(bn * BN + row) * K + k0n + c8) * 2);
        async_load_b128(la, ga, baseA);
        async_load_b128(lb, gb, baseB);
      }
      // 8 async loads/wave in flight for slab s+1; wait out slab s only
      asm volatile("s_wait_asynccnt 0x8" ::: "memory");
    } else {
      asm volatile("s_wait_asynccnt 0x0" ::: "memory");
    }
    __syncthreads();

#pragma unroll
    for (int kk = 0; kk < 2; ++kk) {
      v16bf a0 = load_a_frag(&sA[cur][(rb * 32 + ln) * LDT], grp, kk * 32);
      v16bf a1 = load_a_frag(&sA[cur][(rb * 32 + 16 + ln) * LDT], grp, kk * 32);
#pragma unroll
      for (int t = 0; t < 4; ++t) {
        v16bf bfr = load_b_frag(&sB[cur][(cb * 64 + t * 16 + ln) * LDT], grp, kk * 32);
        acc[0][t] = __builtin_amdgcn_wmma_f32_16x16x32_bf16(
            false, a0, false, bfr, (short)0, acc[0][t], false, false);
        acc[1][t] = __builtin_amdgcn_wmma_f32_16x16x32_bf16(
            false, a1, false, bfr, (short)0, acc[1][t], false, false);
      }
    }
    __syncthreads();   // all waves done with buf[cur] before it is refilled
  }

#pragma unroll
  for (int ms = 0; ms < 2; ++ms)
#pragma unroll
    for (int t = 0; t < 4; ++t)
#pragma unroll
      for (int v = 0; v < 8; ++v) {
        int row = bm * BM + rb * 32 + ms * 16 + v + 8 * grp;
        int col = bn * BN + cb * 64 + t * 16 + ln;
        float val = acc[ms][t][v];
        if (OUT_BF16) Cb[(size_t)row * N + col] = f2bf(val);
        else          Cf[(size_t)row * N + col] = val;
      }
}

// ---------------------------------------------------------------------------
// Flash attention: one block = (b, h, 64-row q tile). 128 threads = 4 waves,
// each wave owns 16 q rows with full d_k=128 accumulation (8 x v8f in VGPRs).
// Q and K tiles staged via async global->LDS; V transposed manually.
// ---------------------------------------------------------------------------
#define QT  64
#define KT  64
#define DKH 128
#define LQ  (DKH + 8)  // 272 B row stride
#define LV  (KT + 8)   // 144 B row stride
#define LSR (KT + 4)
#define LP  (KT + 8)

__global__ __launch_bounds__(128) void flash_attn(const bf16_t* __restrict__ Q,
                                                  const bf16_t* __restrict__ K,
                                                  const bf16_t* __restrict__ V,
                                                  bf16_t* __restrict__ O,
                                                  int Sq, int D, int H) {
  __shared__ bf16_t sQ[QT * LQ];
  __shared__ bf16_t sK[KT * LQ];
  __shared__ bf16_t sVt[DKH * LV];   // transposed: [dim][key]
  __shared__ float  sS[QT * LSR];
  __shared__ bf16_t sP[QT * LP];
  __shared__ float  sM[QT];
  __shared__ float  sL[QT];
  __shared__ float  sAl[QT];

  const int tid  = threadIdx.x;
  const int w    = tid >> 5;
  const int lane = tid & 31;
  const int grp  = lane >> 4;
  const int ln   = lane & 15;

  int       bid = blockIdx.x;
  const int nqt = Sq / QT;
  const int qt  = bid % nqt;  bid /= nqt;
  const int h   = bid % H;
  const int b   = bid / H;
  const int q0  = qt * QT;
  const size_t col0 = (size_t)h * DKH;
  const unsigned long long baseQ = (unsigned long long)Q;
  const unsigned long long baseK = (unsigned long long)K;

  // stage Q tile: 64x128 halves = 1024 16B chunks, 8 per thread (async)
#pragma unroll
  for (int j = 0; j < 8; ++j) {
    int chunk = j * 128 + tid;
    int row   = chunk >> 4;
    int c8    = (chunk & 15) << 3;
    unsigned la = (unsigned)(size_t)&sQ[row * LQ + c8];
    unsigned ga = (unsigned)((((size_t)(b * Sq + q0 + row)) * D + col0 + c8) * 2);
    async_load_b128(la, ga, baseQ);
  }
  if (tid < QT) { sM[tid] = -1e30f; sL[tid] = 0.f; }
  asm volatile("s_wait_asynccnt 0x0" ::: "memory");
  __syncthreads();

  v16bf qa[4];
#pragma unroll
  for (int kk = 0; kk < 4; ++kk)
    qa[kk] = load_a_frag(&sQ[(w * 16 + ln) * LQ], grp, kk * 32);

  v8f acc[8] = {};
  const float rscale = 0.08838834764831844f;  // 1/sqrt(128)

  const int nkt = Sq / KT;
  for (int jt = 0; jt < nkt; ++jt) {
    const int k0 = jt * KT;
    // stage K tile (async) and V tile (manual transpose into sVt)
#pragma unroll
    for (int j = 0; j < 8; ++j) {
      int chunk = j * 128 + tid;
      int row   = chunk >> 4;
      int c8    = (chunk & 15) << 3;
      unsigned la = (unsigned)(size_t)&sK[row * LQ + c8];
      unsigned ga = (unsigned)((((size_t)(b * Sq + k0 + row)) * D + col0 + c8) * 2);
      async_load_b128(la, ga, baseK);
    }
#pragma unroll
    for (int j = 0; j < 16; ++j) {
      int chunk = j * 128 + tid;
      int row   = chunk >> 5;
      int c4    = (chunk & 31) << 2;
      uint2 vv = *(const uint2*)&V[((size_t)(b * Sq + k0 + row)) * D + col0 + c4];
      bf16_t tmp[4];
      __builtin_memcpy(tmp, &vv, 8);
      sVt[(c4 + 0) * LV + row] = tmp[0];
      sVt[(c4 + 1) * LV + row] = tmp[1];
      sVt[(c4 + 2) * LV + row] = tmp[2];
      sVt[(c4 + 3) * LV + row] = tmp[3];
    }
    asm volatile("s_wait_asynccnt 0x0" ::: "memory");
    __syncthreads();

    // S = (Q * K^T) * rscale  -> sS
#pragma unroll
    for (int t = 0; t < 4; ++t) {
      v8f sc = {};
#pragma unroll
      for (int kk = 0; kk < 4; ++kk) {
        v16bf bfr = load_b_frag(&sK[(t * 16 + ln) * LQ], grp, kk * 32);
        sc = __builtin_amdgcn_wmma_f32_16x16x32_bf16(
            false, qa[kk], false, bfr, (short)0, sc, false, false);
      }
#pragma unroll
      for (int v = 0; v < 8; ++v)
        sS[(w * 16 + v + 8 * grp) * LSR + t * 16 + ln] = sc[v] * rscale;
    }
    __syncthreads();

    // online softmax: one thread per q row
    if (tid < QT) {
      float mo = sM[tid];
      float mj = -1e30f;
      for (int c = 0; c < KT; ++c) mj = fmaxf(mj, sS[tid * LSR + c]);
      float mn = fmaxf(mo, mj);
      float al = __expf(mo - mn);
      float sum = 0.f;
      for (int c = 0; c < KT; ++c) {
        float p = __expf(sS[tid * LSR + c] - mn);
        sP[tid * LP + c] = f2bf(p);
        sum += p;
      }
      sL[tid] = sL[tid] * al + sum;
      sM[tid] = mn;
      sAl[tid] = al;
    }
    __syncthreads();

    // rescale running O and accumulate P * V
    float al[8];
#pragma unroll
    for (int v = 0; v < 8; ++v) al[v] = sAl[w * 16 + v + 8 * grp];
#pragma unroll
    for (int t = 0; t < 8; ++t)
#pragma unroll
      for (int v = 0; v < 8; ++v) acc[t][v] *= al[v];

#pragma unroll
    for (int kk = 0; kk < 2; ++kk) {
      v16bf p = load_a_frag(&sP[(w * 16 + ln) * LP], grp, kk * 32);
#pragma unroll
      for (int t = 0; t < 8; ++t) {
        v16bf bfr = load_b_frag(&sVt[(t * 16 + ln) * LV], grp, kk * 32);
        acc[t] = __builtin_amdgcn_wmma_f32_16x16x32_bf16(
            false, p, false, bfr, (short)0, acc[t], false, false);
      }
    }
    __syncthreads();
  }

  float invl[8];
#pragma unroll
  for (int v = 0; v < 8; ++v) invl[v] = 1.0f / sL[w * 16 + v + 8 * grp];
#pragma unroll
  for (int t = 0; t < 8; ++t)
#pragma unroll
    for (int v = 0; v < 8; ++v) {
      int q   = q0 + w * 16 + v + 8 * grp;
      int col = t * 16 + ln;
      O[((size_t)(b * Sq + q)) * D + col0 + col] = f2bf(acc[t][v] * invl[v]);
    }
}

// ---------------------------------------------------------------------------
// host-side launch
// ---------------------------------------------------------------------------
extern "C" void kernel_launch(void* const* d_in, const int* in_sizes, int n_in,
                              void* d_out, int out_size, void* d_ws, size_t ws_size,
                              hipStream_t stream) {
  const int B = 2, S = 2048, D = 2048, H = 16;
  const int M = B * S;                 // 4096 rows
  const size_t nx = (size_t)B * S * D; // 8,388,608
  const size_t nw = (size_t)D * D;     // 4,194,304

  const float* x  = (const float*)d_in[0];
  const float* Wq = (const float*)d_in[1];
  const float* Wk = (const float*)d_in[2];
  const float* Wv = (const float*)d_in[3];
  const float* Wo = (const float*)d_in[4];
  float* out = (float*)d_out;

  char* ws = (char*)d_ws;
  size_t off = 0;
  bf16_t* xb  = (bf16_t*)(ws + off); off += nx * sizeof(bf16_t);
  bf16_t* wqb = (bf16_t*)(ws + off); off += nw * sizeof(bf16_t);
  bf16_t* wkb = (bf16_t*)(ws + off); off += nw * sizeof(bf16_t);
  bf16_t* wvb = (bf16_t*)(ws + off); off += nw * sizeof(bf16_t);
  bf16_t* wob = (bf16_t*)(ws + off); off += nw * sizeof(bf16_t);
  bf16_t* Qb  = (bf16_t*)(ws + off); off += nx * sizeof(bf16_t);
  bf16_t* Kb  = (bf16_t*)(ws + off); off += nx * sizeof(bf16_t);
  bf16_t* Vb  = (bf16_t*)(ws + off); off += nx * sizeof(bf16_t);
  bf16_t* Ob  = (bf16_t*)(ws + off); off += nx * sizeof(bf16_t);

  // 1) fp32 -> bf16
  {
    long n4 = (long)(nx / 4);
    cvt_f32_bf16<<<(unsigned)((n4 + 255) / 256), 256, 0, stream>>>(x, xb, n4);
    long w4 = (long)(nw / 4);
    unsigned gw = (unsigned)((w4 + 255) / 256);
    cvt_f32_bf16<<<gw, 256, 0, stream>>>(Wq, wqb, w4);
    cvt_f32_bf16<<<gw, 256, 0, stream>>>(Wk, wkb, w4);
    cvt_f32_bf16<<<gw, 256, 0, stream>>>(Wv, wvb, w4);
    cvt_f32_bf16<<<gw, 256, 0, stream>>>(Wo, wob, w4);
  }

  // 2) projections Q,K,V = x * W^T   (bf16 out)
  dim3 ggrid(D / BN, M / BM);   // (16, 32)
  gemm_abt<1><<<ggrid, 256, 0, stream>>>(xb, wqb, nullptr, Qb, M, D, D);
  gemm_abt<1><<<ggrid, 256, 0, stream>>>(xb, wkb, nullptr, Kb, M, D, D);
  gemm_abt<1><<<ggrid, 256, 0, stream>>>(xb, wvb, nullptr, Vb, M, D, D);

  // 3) flash attention -> Ob (bf16, [B,S,D] with head-major columns)
  {
    unsigned nblk = (unsigned)(B * H * (S / QT));  // 1024
    flash_attn<<<nblk, 128, 0, stream>>>(Qb, Kb, Vb, Ob, S, D, H);
  }

  // 4) output projection -> d_out (fp32)
  gemm_abt<0><<<ggrid, 256, 0, stream>>>(Ob, wob, out, nullptr, M, D, D);

  (void)in_sizes; (void)n_in; (void)out_size; (void)ws_size;
}
